// LowLevelModel_40106404610844
// MI455X (gfx1250) — matrix operand order repaired and checked by reading, not compiled, not run on hardware
//
#include <hip/hip_runtime.h>
#include <hip/hip_bf16.h>

typedef __attribute__((ext_vector_type(16))) _Float16 v16h;
typedef __attribute__((ext_vector_type(8)))  float    v8f;

#define B    16
#define NAV  36
#define H    120
#define W    160
#define HW   (H*W)          // 19200
#define CHW  (NAV*HW)       // 691200
#define NTD  (B*CHW)        // 11059200
#define HID  512
#define AEMB 32
#define NF   4

#define CH   (3*H)          // 360
#define CW   (12*W)         // 1920
#define C1H  64
#define C1W  269
#define P1H  21
#define P1W  89
#define C2H  10
#define C2W  43

#define KPAD   608          // 584 padded to multiple of 32
#define KS1    19           // K-steps GEMM1 (608/32)
#define KS2    16           // K-steps GEMM2 (512/32)
#define NTILE  128          // 2048/16 N tiles

// workspace layout (float offsets)
#define OFF_NAV   0                                   // 16*36
#define OFF_FATTN (OFF_NAV + B*NAV)                   // 16*4
#define OFF_KERN  (OFF_FATTN + B*NF)                  // 16*36*50 premixed dyn kernels
#define OFF_TD    (OFF_KERN + B*NAV*50)               // td_scaled [16,36,120,160]
#define OFF_CANV  (OFF_TD + NTD)                      // canvas [16,360,1920]
#define OFF_C1    (OFF_CANV + NTD)                    // [2,16,64,269]
#define OFF_P1    (OFF_C1 + 2*B*C1H*C1W)              // [2,16,21,89]
#define OFF_C2    (OFF_P1 + 2*B*P1H*P1W)              // [2,16,10,43]
#define OFF_VF    (OFF_C2 + 2*B*C2H*C2W)              // [16,36]
#define OFF_GATES (OFF_VF + B*NAV)                    // [16,2048]
#define OFF_F16   (OFF_GATES + B*4*HID)               // f16 fragment region (half offsets below)

// half-offsets inside f16 region (all 32B-aligned)
#define HOFF_ASWZ 0                                   // [19][32][16]  A frags, concat
#define HOFF_HSWZ (HOFF_ASWZ + KS1*32*16)             // [16][32][16]  A frags, h0
#define HOFF_BIH  (HOFF_HSWZ + KS2*32*16)             // [128][19][32][16] B frags, w_ih
#define HOFF_BHH  (HOFF_BIH + NTILE*KS1*32*16)        // [128][16][32][16] B frags, w_hh
#define NIH (NTILE*KS1*512)                           // 1245184
#define NHH (NTILE*KS2*512)                           // 1048576

__device__ __forceinline__ float sigf(float x) { return 1.0f / (1.0f + __expf(-x)); }

// A-fragment half index within a 32-wide K chunk, for (lane, element e)
__device__ __forceinline__ int afrag_k(int half, int e) {
    int j = e >> 1;
    return ((j < 4) ? 2*j : 16 + 2*(j-4)) + 8*half + (e & 1);
}

// ---------------- K1: nav softmax + filter-attention MLP + premixed dyn kernels ----
__global__ __launch_bounds__(128) void k_attn(
    const float* __restrict__ numnav, const float* __restrict__ ctx,
    const float* __restrict__ w1, const float* __restrict__ b1,
    const float* __restrict__ w2, const float* __restrict__ b2,
    const float* __restrict__ dynf,
    float* __restrict__ nav_attn, float* __restrict__ filt_attn, float* __restrict__ kern)
{
    const int b = blockIdx.x, t = threadIdx.x;
    __shared__ float hid[128];
    __shared__ float fa[4];
    __shared__ float natt[36];

    if (t == 0) {  // nav softmax (36)
        float mx = -1e30f;
        for (int i = 0; i < NAV; ++i) mx = fmaxf(mx, numnav[b*NAV + i]);
        float sm = 0.f;
        for (int i = 0; i < NAV; ++i) sm += __expf(numnav[b*NAV + i] - mx);
        float inv = 1.f / sm;
        for (int i = 0; i < NAV; ++i) {
            float v = __expf(numnav[b*NAV + i] - mx) * inv;
            natt[i] = v; nav_attn[b*NAV + i] = v;
        }
    }
    // hidden = relu(ctx @ W1 + b1), W1 [512,128]
    float s = b1[t];
    for (int k = 0; k < HID; ++k) s += ctx[b*HID + k] * w1[k*128 + t];
    hid[t] = fmaxf(s, 0.f);
    __syncthreads();
    if (t < NF) {
        float s2 = b2[t];
        for (int k = 0; k < 128; ++k) s2 += hid[k] * w2[k*NF + t];
        fa[t] = s2;
    }
    __syncthreads();
    if (t == 0) {
        float mx = fmaxf(fmaxf(fa[0], fa[1]), fmaxf(fa[2], fa[3]));
        float e0 = __expf(fa[0]-mx), e1 = __expf(fa[1]-mx), e2 = __expf(fa[2]-mx), e3 = __expf(fa[3]-mx);
        float inv = 1.f / (e0+e1+e2+e3);
        fa[0]=e0*inv; fa[1]=e1*inv; fa[2]=e2*inv; fa[3]=e3*inv;
        for (int f = 0; f < NF; ++f) filt_attn[b*NF + f] = fa[f];
    }
    __syncthreads();
    // premix dyn kernels; fold nav attention into c==1 (obj) taps
    for (int u = t; u < NAV*50; u += 128) {
        int g = u / 50, r = u % 50, c = r / 25;
        float v = 0.f;
        for (int f = 0; f < NF; ++f) v += fa[f] * dynf[f*(NAV*50) + g*50 + r];
        if (c == 1) v *= natt[g];
        kern[b*(NAV*50) + u] = v;
    }
}

// ---------------- K2: grouped 5x5 conv (2->1 per group) * nav attention ----------
__global__ __launch_bounds__(256) void k_conv1(
    const float* __restrict__ d0, const float* __restrict__ d1,
    const float* __restrict__ wgt, const float* __restrict__ bias,
    const float* __restrict__ attn, float* __restrict__ td)
{
    int idx = blockIdx.x*256 + threadIdx.x;
    if (idx >= NTD) return;
    int b = idx / CHW, r = idx % CHW;
    int g = r / HW,  p = r % HW;
    int y = p / W,   x = p % W;
    const float* wg  = wgt + g*50;
    const float* in0 = d0 + b*CHW + g*HW;
    const float* in1 = d1 + b*CHW + g*HW;
    float s = bias[g];
    if (y >= 2 && y <= H-3 && x >= 2 && x <= W-3) {        // interior fast path
        const float* p0 = in0 + (y-2)*W + (x-2);
        const float* p1 = in1 + (y-2)*W + (x-2);
        #pragma unroll
        for (int ky = 0; ky < 5; ++ky)
            #pragma unroll
            for (int kx = 0; kx < 5; ++kx)
                s += wg[ky*5+kx] * p0[ky*W+kx] + wg[25+ky*5+kx] * p1[ky*W+kx];
    } else {
        for (int ky = 0; ky < 5; ++ky) {
            int yy = y + ky - 2; if (yy < 0 || yy >= H) continue;
            for (int kx = 0; kx < 5; ++kx) {
                int xx = x + kx - 2; if (xx < 0 || xx >= W) continue;
                int o = yy*W + xx;
                s += wg[ky*5+kx] * in0[o] + wg[25+ky*5+kx] * in1[o];
            }
        }
    }
    td[idx] = s * attn[b*NAV + g];
}

// ---------------- K3: dynamic grouped conv -> write straight into canvas ---------
__global__ __launch_bounds__(256) void k_dynconv(
    const float* __restrict__ td, const float* __restrict__ obj,
    const float* __restrict__ kern, float* __restrict__ canvas)
{
    int idx = blockIdx.x*256 + threadIdx.x;
    if (idx >= NTD) return;
    int b = idx / CHW, r = idx % CHW;
    int g = r / HW,  p = r % HW;
    int y = p / W,   x = p % W;
    const float* kA = kern + b*(NAV*50) + g*50;    // td taps
    const float* kB = kA + 25;                      // obj taps (attn folded in)
    const float* inA = td  + b*CHW + g*HW;
    const float* inB = obj + b*CHW + g*HW;
    float s = 0.f;
    if (y >= 2 && y <= H-3 && x >= 2 && x <= W-3) {        // interior fast path
        const float* pA = inA + (y-2)*W + (x-2);
        const float* pB = inB + (y-2)*W + (x-2);
        #pragma unroll
        for (int ky = 0; ky < 5; ++ky)
            #pragma unroll
            for (int kx = 0; kx < 5; ++kx)
                s += kA[ky*5+kx] * pA[ky*W+kx] + kB[ky*5+kx] * pB[ky*W+kx];
    } else {
        for (int ky = 0; ky < 5; ++ky) {
            int yy = y + ky - 2; if (yy < 0 || yy >= H) continue;
            for (int kx = 0; kx < 5; ++kx) {
                int xx = x + kx - 2; if (xx < 0 || xx >= W) continue;
                int o = yy*W + xx;
                s += kA[ky*5+kx] * inA[o] + kB[ky*5+kx] * inB[o];
            }
        }
    }
    int Y = (2 - g/12)*H + y;
    int X = (g%12)*W + x;
    canvas[b*(CH*CW) + Y*CW + X] = s;
}

// ---------------- K4: conv2_1 (k5, stride 5x7, dil 10) on canvas & rolled --------
__global__ __launch_bounds__(256) void k_conv21(
    const float* __restrict__ canvas, const float* __restrict__ w21,
    const float* __restrict__ b21, float* __restrict__ c1buf)
{
    int idx = blockIdx.x*256 + threadIdx.x;
    if (idx >= 2*B*C1H*C1W) return;
    int v  = idx / (B*C1H*C1W), r = idx % (B*C1H*C1W);
    int b  = r / (C1H*C1W),     q = r % (C1H*C1W);
    int oy = q / C1W,           ox = q % C1W;
    float s = b21[0];
    #pragma unroll
    for (int ky = 0; ky < 5; ++ky) {
        int Y = 5*oy + 10*ky;
        #pragma unroll
        for (int kx = 0; kx < 5; ++kx) {
            int Xr = 7*ox + 10*kx;
            int X = v ? (Xr + (CW - W)) % CW : Xr;   // rolled[x] = canvas[(x-160) mod 1920]
            s += w21[ky*5+kx] * canvas[b*(CH*CW) + Y*CW + X];
        }
    }
    c1buf[idx] = s;
}

// ---------------- K5: 3x3/3 avgpool --------------------------------------------
__global__ __launch_bounds__(256) void k_pool1(const float* __restrict__ c1buf, float* __restrict__ p1buf)
{
    int idx = blockIdx.x*256 + threadIdx.x;
    if (idx >= 2*B*P1H*P1W) return;
    int vb = idx / (P1H*P1W), q = idx % (P1H*P1W);
    int py = q / P1W, px = q % P1W;
    float s = 0.f;
    #pragma unroll
    for (int i = 0; i < 3; ++i)
        #pragma unroll
        for (int j = 0; j < 3; ++j)
            s += c1buf[vb*(C1H*C1W) + (3*py+i)*C1W + (3*px+j)];
    p1buf[idx] = s * (1.f/9.f);
}

// ---------------- K6: conv2_2 (k3, stride 2, dil (1,2)) --------------------------
__global__ __launch_bounds__(256) void k_conv22(
    const float* __restrict__ p1buf, const float* __restrict__ w22,
    const float* __restrict__ b22, float* __restrict__ c2buf)
{
    int idx = blockIdx.x*256 + threadIdx.x;
    if (idx >= 2*B*C2H*C2W) return;
    int vb = idx / (C2H*C2W), q = idx % (C2H*C2W);
    int oy = q / C2W, ox = q % C2W;
    float s = b22[0];
    #pragma unroll
    for (int ky = 0; ky < 3; ++ky)
        #pragma unroll
        for (int kx = 0; kx < 3; ++kx)
            s += w22[ky*3+kx] * p1buf[vb*(P1H*P1W) + (2*oy+ky)*P1W + (2*ox + 2*kx)];
    c2buf[idx] = s;
}

// ---------------- K7: (3,9)/3 avgpool + roll + flip + softmax(/10) ---------------
__global__ __launch_bounds__(64) void k_poolvf(
    const float* __restrict__ c2buf, float* __restrict__ vf, float* __restrict__ out_vf)
{
    int b = blockIdx.x, t = threadIdx.x;
    __shared__ float lds[NAV];
    if (t < NAV) {
        int rr = t / 12, c = t % 12;
        int srow = 2 - rr;                 // row flip
        float s = 0.f;
        #pragma unroll
        for (int i = 0; i < 3; ++i)
            for (int j = 0; j < 9; ++j) {
                s += c2buf[(0*B + b)*(C2H*C2W) + (3*srow+i)*C2W + (3*c + j)];
                s += c2buf[(1*B + b)*(C2H*C2W) + (3*srow+i)*C2W + (3*((c+1)%12) + j)];
            }
        lds[t] = (s * (1.f/27.f)) * 0.5f * 0.1f;   // (v1+v2)/2, /10 temperature
    }
    __syncthreads();
    if (t < NAV) {
        float mx = -1e30f;
        for (int i = 0; i < NAV; ++i) mx = fmaxf(mx, lds[i]);
        float sm = 0.f;
        for (int i = 0; i < NAV; ++i) sm += __expf(lds[i] - mx);
        float p = __expf(lds[t] - mx) / sm;
        vf[b*NAV + t] = p;
        out_vf[b*NAV + t] = p;
    }
}

// ---------------- K8: pack A fragments (concat + h0) in WMMA lane order ----------
// aswz[kstep][lane][e], hswz[kstep][lane][e]; value(m,k) from concat sources.
__global__ __launch_bounds__(256) void k_aswz(
    const float* __restrict__ ctx, const float* __restrict__ vf,
    const float* __restrict__ fattn, const float* __restrict__ pre,
    const float* __restrict__ h0,
    _Float16* __restrict__ aswz, _Float16* __restrict__ hswz)
{
    int idx = blockIdx.x*256 + threadIdx.x;
    if (idx < KS1*512) {
        int kstep = idx / 512, q = idx % 512;
        int lane = q / 16, e = q % 16;
        int half = lane >> 4, m = lane & 15;
        int k = kstep*32 + afrag_k(half, e);
        float v;
        if      (k < HID)             v = ctx[m*HID + k];
        else if (k < HID+NAV)         v = vf[m*NAV + (k-HID)];
        else if (k < HID+NAV+NF)      v = fattn[m*NF + (k-HID-NAV)];
        else if (k < HID+NAV+NF+AEMB) v = pre[m*AEMB + (k-HID-NAV-NF)];
        else                          v = 0.f;
        aswz[idx] = (_Float16)v;
    } else if (idx < KS1*512 + KS2*512) {
        int t = idx - KS1*512;
        int kstep = t / 512, q = t % 512;
        int lane = q / 16, e = q % 16;
        int half = lane >> 4, m = lane & 15;
        int k = kstep*32 + afrag_k(half, e);
        hswz[t] = (_Float16)h0[m*HID + k];
    }
}

// ---------------- K9: pack B fragments (weights, f32->f16, zero-pad K) -----------
// bih[tile][kstep][lane][e] = w_ih[n, k], n = tile*16 + (lane&15), k = kstep*32 + afrag_k(lane>>4, e)
__global__ __launch_bounds__(256) void k_wcvt(
    const float* __restrict__ w_ih, const float* __restrict__ w_hh,
    _Float16* __restrict__ bih, _Float16* __restrict__ bhh)
{
    int idx = blockIdx.x*256 + threadIdx.x;
    if (idx < NIH) {
        int tile = idx / (KS1*512), r = idx % (KS1*512);
        int kstep = r / 512, q = r % 512;
        int lane = q / 16, e = q % 16;
        int n = tile*16 + (lane & 15);
        int k = kstep*32 + afrag_k(lane >> 4, e);
        bih[idx] = (k < 584) ? (_Float16)w_ih[n*584 + k] : (_Float16)0.0f;
    } else if (idx < NIH + NHH) {
        int t = idx - NIH;
        int tile = t / (KS2*512), r = t % (KS2*512);
        int kstep = r / 512, q = r % 512;
        int lane = q / 16, e = q % 16;
        int n = tile*16 + (lane & 15);
        int k = kstep*32 + afrag_k(lane >> 4, e);
        bhh[t] = (_Float16)w_hh[n*HID + k];
    }
}

// ---------------- K10: LSTM gates GEMM, pure fragment-load + WMMA loop -----------
__global__ __launch_bounds__(256) void k_lstm_wmma(
    const _Float16* __restrict__ aswz, const _Float16* __restrict__ hswz,
    const _Float16* __restrict__ bih, const _Float16* __restrict__ bhh,
    const float* __restrict__ b_ih, const float* __restrict__ b_hh,
    float* __restrict__ gates)
{
    const int lane = threadIdx.x & 31;
    const int wave = threadIdx.x >> 5;
    const int tile = blockIdx.x * 8 + wave;      // 0..127
    const int half = lane >> 4;
    const int ncol = tile*16 + (lane & 15);

    const v16h* A1 = (const v16h*)aswz;                      // [19][32]
    const v16h* A2 = (const v16h*)hswz;                      // [16][32]
    const v16h* B1 = (const v16h*)bih + tile*(KS1*32);       // [19][32]
    const v16h* B2 = (const v16h*)bhh + tile*(KS2*32);       // [16][32]

    v8f acc = {};
    #pragma unroll 4
    for (int s = 0; s < KS1; ++s) {
        v16h a = A1[s*32 + lane];
        v16h b = B1[s*32 + lane];
        acc = __builtin_amdgcn_wmma_f32_16x16x32_f16(false, a, false, b, (short)0, acc, false, false);
    }
    #pragma unroll 4
    for (int s = 0; s < KS2; ++s) {
        v16h a = A2[s*32 + lane];
        v16h b = B2[s*32 + lane];
        acc = __builtin_amdgcn_wmma_f32_16x16x32_f16(false, a, false, b, (short)0, acc, false, false);
    }
    const float bias = b_ih[ncol] + b_hh[ncol];
    #pragma unroll
    for (int j = 0; j < 8; ++j) {                 // C/D layout: VGPR j -> m = j + 8*half
        int m = j + 8*half;
        gates[m*2048 + ncol] = acc[j] + bias;
    }
}

// ---------------- K11: gate nonlinearity -> h_1, c_1 ----------------------------
__global__ __launch_bounds__(256) void k_gates(
    const float* __restrict__ gates, const float* __restrict__ c0, float* __restrict__ out)
{
    int idx = blockIdx.x*256 + threadIdx.x;
    if (idx >= B*HID) return;
    int m = idx / HID, u = idx % HID;
    const float* g = gates + m*2048;
    float gi = g[u], gf = g[HID + u], gg = g[2*HID + u], go = g[3*HID + u];
    float c1 = sigf(gf) * c0[idx] + sigf(gi) * tanhf(gg);
    float h1 = sigf(go) * tanhf(c1);
    out[idx] = h1;                 // h_1
    out[B*HID + idx] = c1;         // c_1
}

// ---------------- K12: navigability mask ----------------------------------------
__global__ __launch_bounds__(16) void k_mask(const int* __restrict__ navidx, float* __restrict__ out_mask)
{
    int b = threadIdx.x;
    if (b >= B) return;
    float* row = out_mask + b*(NAV+1);
    for (int j = 0; j < NAV+1; ++j) row[j] = 0.f;
    row[0] = 1.f;
    for (int i = 0; i < 8; ++i) row[navidx[b*8 + i] + 1] = 1.f;
}

extern "C" void kernel_launch(void* const* d_in, const int* in_sizes, int n_in,
                              void* d_out, int out_size, void* d_ws, size_t ws_size,
                              hipStream_t stream) {
    const float* depth0  = (const float*)d_in[0];
    const float* depth1  = (const float*)d_in[1];
    const float* obj     = (const float*)d_in[2];
    const float* numnav  = (const float*)d_in[3];
    const float* pre     = (const float*)d_in[4];
    const float* h0      = (const float*)d_in[5];
    const float* c0      = (const float*)d_in[6];
    const float* ctx     = (const float*)d_in[7];
    const int*   navidx  = (const int*)d_in[8];
    const float* conv1_w = (const float*)d_in[9];
    const float* conv1_b = (const float*)d_in[10];
    const float* dynf    = (const float*)d_in[11];
    const float* dyn_w1  = (const float*)d_in[12];
    const float* dyn_b1  = (const float*)d_in[13];
    const float* dyn_w2  = (const float*)d_in[14];
    const float* dyn_b2  = (const float*)d_in[15];
    const float* w21     = (const float*)d_in[16];
    const float* b21     = (const float*)d_in[17];
    const float* w22     = (const float*)d_in[18];
    const float* b22     = (const float*)d_in[19];
    const float* w_ih    = (const float*)d_in[20];
    const float* w_hh    = (const float*)d_in[21];
    const float* b_ih    = (const float*)d_in[22];
    const float* b_hh    = (const float*)d_in[23];

    float* ws      = (float*)d_ws;
    float* nav     = ws + OFF_NAV;
    float* fattn   = ws + OFF_FATTN;
    float* kern    = ws + OFF_KERN;
    float* td      = ws + OFF_TD;
    float* canvas  = ws + OFF_CANV;
    float* c1buf   = ws + OFF_C1;
    float* p1buf   = ws + OFF_P1;
    float* c2buf   = ws + OFF_C2;
    float* vf      = ws + OFF_VF;
    float* gates   = ws + OFF_GATES;
    _Float16* f16r = (_Float16*)(ws + OFF_F16);
    _Float16* aswz = f16r + HOFF_ASWZ;
    _Float16* hswz = f16r + HOFF_HSWZ;
    _Float16* bih  = f16r + HOFF_BIH;
    _Float16* bhh  = f16r + HOFF_BHH;

    float* out = (float*)d_out;
    float* out_vf   = out + 2*B*HID;           // 16384
    float* out_mask = out + 2*B*HID + B*NAV;   // 16960

    // weight pack is independent of the pipeline: launch first
    k_wcvt<<<(NIH + NHH + 255)/256, 256, 0, stream>>>(w_ih, w_hh, bih, bhh);
    k_attn<<<B, 128, 0, stream>>>(numnav, ctx, dyn_w1, dyn_b1, dyn_w2, dyn_b2, dynf,
                                  nav, fattn, kern);
    k_conv1<<<(NTD + 255)/256, 256, 0, stream>>>(depth0, depth1, conv1_w, conv1_b, nav, td);
    k_dynconv<<<(NTD + 255)/256, 256, 0, stream>>>(td, obj, kern, canvas);
    k_conv21<<<(2*B*C1H*C1W + 255)/256, 256, 0, stream>>>(canvas, w21, b21, c1buf);
    k_pool1<<<(2*B*P1H*P1W + 255)/256, 256, 0, stream>>>(c1buf, p1buf);
    k_conv22<<<(2*B*C2H*C2W + 255)/256, 256, 0, stream>>>(p1buf, w22, b22, c2buf);
    k_poolvf<<<B, 64, 0, stream>>>(c2buf, vf, out_vf);
    k_aswz<<<((KS1+KS2)*512 + 255)/256, 256, 0, stream>>>(ctx, vf, fattn, pre, h0, aswz, hswz);
    k_lstm_wmma<<<16, 256, 0, stream>>>(aswz, hswz, bih, bhh, b_ih, b_hh, gates);
    k_gates<<<(B*HID + 255)/256, 256, 0, stream>>>(gates, c0, out);
    k_mask<<<1, 16, 0, stream>>>(navidx, out_mask);
}
